// EmbeddingModule_41102837023000
// MI455X (gfx1250) — matrix operand order repaired and checked by reading, not compiled, not run on hardware
//
#include <hip/hip_runtime.h>
#include <math.h>

// ---------------------------------------------------------------------------
// MI455X (gfx1250) implementation of the 2-layer full-attention embedding
// module. All GEMMs (input proj, QKV proj, Q@K^T, P@V) run through
// v_wmma_f32_16x16x32_bf16 (bf16 operands, fp32 accumulation).
//
// Data-movement strategy:
//  * Weights are transposed+converted ONCE to bf16 [n][k] (k_wt), so every
//    GEMM B-tile is a contiguous row copy -> GLOBAL_LOAD_ASYNC_TO_LDS_B128.
//  * The V projection writes its output directly transposed [HD][NN], so the
//    P@V B-tiles are also contiguous async copies.
//  * Attention is two-pass: (1) S = scale*Q@K^T stored fp32 + online row
//    max/denominator per 64-row strip (no atomics); (2) O = P@V with
//    P = exp(S-m)*invl generated during A-tile LDS staging.
//  * All pure bf16 tile copies use gfx1250 async global->LDS (ASYNCcnt),
//    overlapping with WMMA compute on the other LDS buffer.
// edge_indices / edge_times / timestamps / nodes are dead in the reference.
// ---------------------------------------------------------------------------

typedef __bf16 bf16;
typedef __attribute__((ext_vector_type(16))) __bf16 v16bf;
typedef __attribute__((ext_vector_type(8)))  __bf16 v8bf;
typedef __attribute__((ext_vector_type(8)))  float  v8f;
typedef __attribute__((ext_vector_type(4)))  float  v4f;

#define NN    8192        // nodes
#define FDIM  256         // feature dim (== memory dim)
#define KIN   512         // F+M
#define HD    1024        // hidden
#define STR   40          // LDS row stride in bf16 elems (32 + 8 pad -> 80B)
#define SCALE 0.03125f    // 1/sqrt(1024)

union FragBF { v16bf v; v8bf h[2]; };

__device__ __forceinline__ v8f wmma_bf16(v16bf a, v16bf b, v8f c) {
  return __builtin_amdgcn_wmma_f32_16x16x32_bf16(
      false, a, false, b, (short)0, c, false, false);
}

// gfx1250 async global->LDS copy, 16B per lane. ldsaddr = LDS byte offset
// (low 32 bits of the generic pointer, per flat-aperture addressing).
__device__ __forceinline__ void async_ld_b128(const bf16* lds_dst,
                                              const bf16* gsrc) {
  unsigned ldsaddr = (unsigned)(unsigned long long)lds_dst;
  unsigned long long gaddr = (unsigned long long)gsrc;
  asm volatile("global_load_async_to_lds_b128 %0, %1, off"
               :: "v"(ldsaddr), "v"(gaddr) : "memory");
}
__device__ __forceinline__ void wait_async0() {
  asm volatile("s_wait_asynccnt 0x0" ::: "memory");
}

// A fragment (16x32 bf16): lane m=lane&15, half=lane>>4.
// VGPR0-3 hold K = half*8 + 0..7, VGPR4-7 hold K = 16 + half*8 + 0..7.
// LDS tile is row-major [row][k], row stride STR.
__device__ __forceinline__ v16bf frag_a(const bf16* t, int lane, int row0) {
  int m = lane & 15, half = lane >> 4;
  FragBF f;
  const bf16* p = t + (row0 + m) * STR + half * 8;
  f.h[0] = *(const v8bf*)(p);
  f.h[1] = *(const v8bf*)(p + 16);
  return f.v;
}

// B fragment (32x16 bf16): lane n=lane&15, half=lane>>4.
// VGPR0-3 hold K = half*8 + 0..7, VGPR4-7 hold K = 16 + half*8 + 0..7.
// LDS tile is transposed [n][k], row stride STR -> contiguous per lane.
__device__ __forceinline__ v16bf frag_b(const bf16* t, int lane, int col0) {
  int n = lane & 15, half = lane >> 4;
  FragBF f;
  const bf16* p = t + (col0 + n) * STR + half * 8;
  f.h[0] = *(const v8bf*)(p);
  f.h[1] = *(const v8bf*)(p + 16);
  return f.v;
}

__device__ __forceinline__ float redmax16(float v) {
  v = fmaxf(v, __shfl_xor(v, 1, 32));
  v = fmaxf(v, __shfl_xor(v, 2, 32));
  v = fmaxf(v, __shfl_xor(v, 4, 32));
  v = fmaxf(v, __shfl_xor(v, 8, 32));
  return v;
}
__device__ __forceinline__ float redsum16(float v) {
  v += __shfl_xor(v, 1, 32);
  v += __shfl_xor(v, 2, 32);
  v += __shfl_xor(v, 4, 32);
  v += __shfl_xor(v, 8, 32);
  return v;
}

// ---- staging helpers ------------------------------------------------------

// 128 rows x 32 cols bf16 rows -> LDS row-major [r][k], async (256 threads).
// Used for GEMM A-tiles, pre-transposed weight B-tiles, and Vt B-tiles.
__device__ __forceinline__ void stage128_async(const bf16* dst,
                                               const bf16* src, int ld,
                                               int row0, int k0, int tid) {
#pragma unroll
  for (int it = 0; it < 2; ++it) {
    int s = tid + it * 256;
    int r = s >> 2, c8 = (s & 3) * 8;
    async_ld_b128(dst + r * STR + c8,
                  src + (size_t)(row0 + r) * ld + k0 + c8);
  }
}

// 64 rows x 32 cols bf16 rows -> LDS row-major, async (128 threads).
__device__ __forceinline__ void stage64_async(const bf16* dst,
                                              const bf16* src,
                                              int row0, int k0, int tid) {
#pragma unroll
  for (int it = 0; it < 2; ++it) {
    int s = tid + it * 128;
    int r = s >> 2, c8 = (s & 3) * 8;
    async_ld_b128(dst + r * STR + c8,
                  src + (size_t)(row0 + r) * HD + k0 + c8);
  }
}

// concat(features, memory) f32 tile 128r x 32k -> bf16 LDS row-major.
__device__ __forceinline__ void stage_cat(bf16* dst, const float* F,
                                          const float* M, int row0, int k0,
                                          int tid) {
  v4f w[4];
#pragma unroll
  for (int it = 0; it < 4; ++it) {
    int s = tid + it * 256;
    int r = s >> 3, c4 = (s & 7) * 4;
    int gk = k0 + c4;
    const float* base = (gk < FDIM)
        ? (F + (size_t)(row0 + r) * FDIM + gk)
        : (M + (size_t)(row0 + r) * FDIM + (gk - FDIM));
    w[it] = *(const v4f*)base;
  }
#pragma unroll
  for (int it = 0; it < 4; ++it) {
    int s = tid + it * 256;
    int r = s >> 3, c4 = (s & 7) * 4;
    bf16* d = dst + r * STR + c4;
    d[0] = (bf16)w[it][0]; d[1] = (bf16)w[it][1];
    d[2] = (bf16)w[it][2]; d[3] = (bf16)w[it][3];
  }
}

// P = exp(S - m) * invl : S f32 tile 128r x 32k -> bf16 LDS row-major.
__device__ __forceinline__ void stage_p(bf16* dst, const float* S,
                                        const float* Mst, const float* Li,
                                        int row0, int k0, int tid) {
  v4f sv[4];
  float m[4], il[4];
#pragma unroll
  for (int it = 0; it < 4; ++it) {
    int s = tid + it * 256;
    int r = s >> 3, c4 = (s & 7) * 4;
    int row = row0 + r;
    m[it] = Mst[row];
    il[it] = Li[row];
    sv[it] = *(const v4f*)(S + (size_t)row * NN + k0 + c4);
  }
#pragma unroll
  for (int it = 0; it < 4; ++it) {
    int s = tid + it * 256;
    int r = s >> 3, c4 = (s & 7) * 4;
    bf16* d = dst + r * STR + c4;
    d[0] = (bf16)(__expf(sv[it][0] - m[it]) * il[it]);
    d[1] = (bf16)(__expf(sv[it][1] - m[it]) * il[it]);
    d[2] = (bf16)(__expf(sv[it][2] - m[it]) * il[it]);
    d[3] = (bf16)(__expf(sv[it][3] - m[it]) * il[it]);
  }
}

// ---- one-time weight transpose: W f32 [K][HD] -> Wt bf16 [HD][K] ----------
__global__ __launch_bounds__(256)
void k_wt(const float* __restrict__ W, bf16* __restrict__ Wt, int K) {
  __shared__ bf16 t[32][33];
  int k0 = blockIdx.x * 32, n0 = blockIdx.y * 32;
  int tx = threadIdx.x & 31, ty = threadIdx.x >> 5;  // 32 x 8
#pragma unroll
  for (int i = 0; i < 4; i++) {
    int k = k0 + ty + i * 8;
    t[ty + i * 8][tx] = (bf16)W[(size_t)k * HD + n0 + tx];
  }
  __syncthreads();
#pragma unroll
  for (int i = 0; i < 4; i++) {
    int n = n0 + ty + i * 8;
    Wt[(size_t)n * K + k0 + tx] = t[tx][ty + i * 8];
  }
}

// ---- input projection: emb = concat(F,M) @ W_in + b  (writes f32 + bf16) --
__global__ __launch_bounds__(256)
void k_inproj(const float* __restrict__ F, const float* __restrict__ M,
              const bf16* __restrict__ Wt, const float* __restrict__ bias,
              float* __restrict__ Ef, bf16* __restrict__ Eb) {
  __shared__ bf16 sA[2][128 * STR];
  __shared__ bf16 sB[2][128 * STR];
  int tid = threadIdx.x, lane = tid & 31, wid = tid >> 5;
  int wm = wid >> 2, wn = wid & 3;
  int m0 = blockIdx.x * 128, n0 = blockIdx.y * 128;
  const int nsteps = KIN / 32;

  v8f z = {0, 0, 0, 0, 0, 0, 0, 0};
  v8f acc[4][2];
#pragma unroll
  for (int i = 0; i < 4; i++) for (int j = 0; j < 2; j++) acc[i][j] = z;

  stage_cat(sA[0], F, M, m0, 0, tid);
  stage128_async(sB[0], Wt, KIN, n0, 0, tid);
  wait_async0();
  __syncthreads();
  for (int s = 0; s < nsteps; s++) {
    int cur = s & 1;
    if (s + 1 < nsteps) {
      stage_cat(sA[cur ^ 1], F, M, m0, (s + 1) * 32, tid);
      stage128_async(sB[cur ^ 1], Wt, KIN, n0, (s + 1) * 32, tid);
    }
    v16bf bfr[2];
#pragma unroll
    for (int j = 0; j < 2; j++) bfr[j] = frag_b(sB[cur], lane, wn * 32 + j * 16);
#pragma unroll
    for (int i = 0; i < 4; i++) {
      v16bf af = frag_a(sA[cur], lane, wm * 64 + i * 16);
#pragma unroll
      for (int j = 0; j < 2; j++) acc[i][j] = wmma_bf16(af, bfr[j], acc[i][j]);
    }
    wait_async0();
    __syncthreads();
  }
  int n = lane & 15, half = lane >> 4;
#pragma unroll
  for (int j = 0; j < 2; j++) {
    int gc = n0 + wn * 32 + j * 16 + n;
    float bv = bias[gc];
#pragma unroll
    for (int i = 0; i < 4; i++)
#pragma unroll
      for (int r = 0; r < 8; r++) {
        int gr = m0 + wm * 64 + i * 16 + half * 8 + r;
        float val = acc[i][j][r] + bv;
        Ef[(size_t)gr * HD + gc] = val;
        Eb[(size_t)gr * HD + gc] = (bf16)val;
      }
  }
}

// ---- Q/K/V projection: out = emb_bf16 @ W + b (bf16 out) ------------------
// TROUT=false: row-major [NN][HD] (q,k). TROUT=true: transposed [HD][NN] (v),
// per-lane stores are 8 consecutive rows at a fixed column -> contiguous 16B.
template <bool TROUT>
__global__ __launch_bounds__(256)
void k_proj(const bf16* __restrict__ A, const bf16* __restrict__ Wt,
            const float* __restrict__ bias, bf16* __restrict__ O) {
  __shared__ bf16 sA[2][128 * STR];
  __shared__ bf16 sB[2][128 * STR];
  int tid = threadIdx.x, lane = tid & 31, wid = tid >> 5;
  int wm = wid >> 2, wn = wid & 3;
  int m0 = blockIdx.x * 128, n0 = blockIdx.y * 128;
  const int nsteps = HD / 32;

  v8f z = {0, 0, 0, 0, 0, 0, 0, 0};
  v8f acc[4][2];
#pragma unroll
  for (int i = 0; i < 4; i++) for (int j = 0; j < 2; j++) acc[i][j] = z;

  stage128_async(sA[0], A, HD, m0, 0, tid);
  stage128_async(sB[0], Wt, HD, n0, 0, tid);
  wait_async0();
  __syncthreads();
  for (int s = 0; s < nsteps; s++) {
    int cur = s & 1;
    if (s + 1 < nsteps) {
      stage128_async(sA[cur ^ 1], A, HD, m0, (s + 1) * 32, tid);
      stage128_async(sB[cur ^ 1], Wt, HD, n0, (s + 1) * 32, tid);
    }
    v16bf bfr[2];
#pragma unroll
    for (int j = 0; j < 2; j++) bfr[j] = frag_b(sB[cur], lane, wn * 32 + j * 16);
#pragma unroll
    for (int i = 0; i < 4; i++) {
      v16bf af = frag_a(sA[cur], lane, wm * 64 + i * 16);
#pragma unroll
      for (int j = 0; j < 2; j++) acc[i][j] = wmma_bf16(af, bfr[j], acc[i][j]);
    }
    wait_async0();
    __syncthreads();
  }
  int n = lane & 15, half = lane >> 4;
#pragma unroll
  for (int j = 0; j < 2; j++) {
    int gc = n0 + wn * 32 + j * 16 + n;
    float bv = bias[gc];
#pragma unroll
    for (int i = 0; i < 4; i++)
#pragma unroll
      for (int r = 0; r < 8; r++) {
        int gr = m0 + wm * 64 + i * 16 + half * 8 + r;
        if (TROUT)
          O[(size_t)gc * NN + gr] = (bf16)(acc[i][j][r] + bv);
        else
          O[(size_t)gr * HD + gc] = (bf16)(acc[i][j][r] + bv);
      }
  }
}

// ---- pass 1: S = scale*Q@K^T (fp32 to HBM) + online row max / denom -------
// One 128-thread block (4 waves) owns a 64-row strip; each wave 16 rows.
__global__ __launch_bounds__(128)
void k_scores(const bf16* __restrict__ Q, const bf16* __restrict__ Km,
              float* __restrict__ S, float* __restrict__ Mst,
              float* __restrict__ Li) {
  __shared__ bf16 sQ[2][64 * STR];
  __shared__ bf16 sK[2][64 * STR];
  int tid = threadIdx.x, lane = tid & 31, wid = tid >> 5;
  int row0 = blockIdx.x * 64;
  int n = lane & 15, half = lane >> 4;

  float m_run[8], l_run[8];
#pragma unroll
  for (int r = 0; r < 8; r++) { m_run[r] = -INFINITY; l_run[r] = 0.f; }
  v8f z = {0, 0, 0, 0, 0, 0, 0, 0};

  for (int n0 = 0; n0 < NN; n0 += 64) {
    v8f acc[4];
#pragma unroll
    for (int j = 0; j < 4; j++) acc[j] = z;

    stage64_async(sQ[0], Q, row0, 0, tid);
    stage64_async(sK[0], Km, n0, 0, tid);
    wait_async0();
    __syncthreads();
    for (int s = 0; s < HD / 32; s++) {
      int cur = s & 1;
      if (s + 1 < HD / 32) {
        stage64_async(sQ[cur ^ 1], Q, row0, (s + 1) * 32, tid);
        stage64_async(sK[cur ^ 1], Km, n0, (s + 1) * 32, tid);
      }
      v16bf af = frag_a(sQ[cur], lane, wid * 16);
#pragma unroll
      for (int j = 0; j < 4; j++)
        acc[j] = wmma_bf16(af, frag_b(sK[cur], lane, j * 16), acc[j]);
      wait_async0();
      __syncthreads();
    }
    // scale, online softmax stats, store S
#pragma unroll
    for (int j = 0; j < 4; j++)
#pragma unroll
      for (int r = 0; r < 8; r++) acc[j][r] *= SCALE;
#pragma unroll
    for (int r = 0; r < 8; r++) {
      float mx = fmaxf(fmaxf(acc[0][r], acc[1][r]), fmaxf(acc[2][r], acc[3][r]));
      mx = redmax16(mx);
      float mnew = fmaxf(m_run[r], mx);
      float ps = __expf(acc[0][r] - mnew) + __expf(acc[1][r] - mnew) +
                 __expf(acc[2][r] - mnew) + __expf(acc[3][r] - mnew);
      ps = redsum16(ps);
      l_run[r] = l_run[r] * __expf(m_run[r] - mnew) + ps;
      m_run[r] = mnew;
    }
    int grb = row0 + wid * 16 + half * 8;
#pragma unroll
    for (int j = 0; j < 4; j++)
#pragma unroll
      for (int r = 0; r < 8; r++)
        S[(size_t)(grb + r) * NN + n0 + j * 16 + n] = acc[j][r];
    __syncthreads();
  }
  if (n == 0) {  // lanes 0 and 16: one writer per row
    int grb = row0 + wid * 16 + half * 8;
#pragma unroll
    for (int r = 0; r < 8; r++) {
      Mst[grb + r] = m_run[r];
      Li[grb + r] = 1.0f / l_run[r];
    }
  }
}

// ---- pass 2: O = P@V, P = exp(S-m)*invl generated in A-staging ------------
// B-tiles come from the pre-transposed V (Vt: [HD][NN] bf16) -> async copy.
__global__ __launch_bounds__(256)
void k_pv(const float* __restrict__ S, const float* __restrict__ Mst,
          const float* __restrict__ Li, const bf16* __restrict__ Vt,
          float* __restrict__ O) {
  __shared__ bf16 sA[2][128 * STR];
  __shared__ bf16 sB[2][128 * STR];
  int tid = threadIdx.x, lane = tid & 31, wid = tid >> 5;
  int wm = wid >> 2, wn = wid & 3;
  int m0 = blockIdx.x * 128, n0 = blockIdx.y * 128;
  const int nsteps = NN / 32;

  v8f z = {0, 0, 0, 0, 0, 0, 0, 0};
  v8f acc[4][2];
#pragma unroll
  for (int i = 0; i < 4; i++) for (int j = 0; j < 2; j++) acc[i][j] = z;

  stage_p(sA[0], S, Mst, Li, m0, 0, tid);
  stage128_async(sB[0], Vt, NN, n0, 0, tid);
  wait_async0();
  __syncthreads();
  for (int s = 0; s < nsteps; s++) {
    int cur = s & 1;
    if (s + 1 < nsteps) {
      stage_p(sA[cur ^ 1], S, Mst, Li, m0, (s + 1) * 32, tid);
      stage128_async(sB[cur ^ 1], Vt, NN, n0, (s + 1) * 32, tid);
    }
    v16bf bfr[2];
#pragma unroll
    for (int j = 0; j < 2; j++) bfr[j] = frag_b(sB[cur], lane, wn * 32 + j * 16);
#pragma unroll
    for (int i = 0; i < 4; i++) {
      v16bf af = frag_a(sA[cur], lane, wm * 64 + i * 16);
#pragma unroll
      for (int j = 0; j < 2; j++) acc[i][j] = wmma_bf16(af, bfr[j], acc[i][j]);
    }
    wait_async0();
    __syncthreads();
  }
  int n = lane & 15, half = lane >> 4;
#pragma unroll
  for (int j = 0; j < 2; j++) {
    int gc = n0 + wn * 32 + j * 16 + n;
#pragma unroll
    for (int i = 0; i < 4; i++)
#pragma unroll
      for (int r = 0; r < 8; r++) {
        int gr = m0 + wm * 64 + i * 16 + half * 8 + r;
        O[(size_t)gr * HD + gc] = acc[i][j][r];
      }
  }
}

// ---- residual + LayerNorm: emb = LN(emb + att)*g + b (writes f32 + bf16) --
__global__ __launch_bounds__(256)
void k_ln(const float* __restrict__ X, const float* __restrict__ Att,
          const float* __restrict__ g, const float* __restrict__ b,
          float* __restrict__ Of, bf16* __restrict__ Ob) {
  __shared__ float red[2][8];
  int row = blockIdx.x, tid = threadIdx.x, lane = tid & 31, wid = tid >> 5;
  float v[4], s = 0.f, s2 = 0.f;
#pragma unroll
  for (int i = 0; i < 4; i++) {
    int c = tid + i * 256;
    float x = X[(size_t)row * HD + c] + Att[(size_t)row * HD + c];
    v[i] = x; s += x; s2 += x * x;
  }
  for (int mask = 1; mask < 32; mask <<= 1) {
    s += __shfl_xor(s, mask, 32);
    s2 += __shfl_xor(s2, mask, 32);
  }
  if (lane == 0) { red[0][wid] = s; red[1][wid] = s2; }
  __syncthreads();
  float ts = 0.f, ts2 = 0.f;
#pragma unroll
  for (int w = 0; w < 8; w++) { ts += red[0][w]; ts2 += red[1][w]; }
  float mean = ts * (1.0f / HD);
  float var = ts2 * (1.0f / HD) - mean * mean;
  float rs = rsqrtf(var + 1e-5f);
#pragma unroll
  for (int i = 0; i < 4; i++) {
    int c = tid + i * 256;
    float y = (v[i] - mean) * rs * g[c] + b[c];
    Of[(size_t)row * HD + c] = y;
    Ob[(size_t)row * HD + c] = (bf16)y;
  }
}

// ---------------------------------------------------------------------------
extern "C" void kernel_launch(void* const* d_in, const int* in_sizes, int n_in,
                              void* d_out, int out_size, void* d_ws,
                              size_t ws_size, hipStream_t stream) {
  const float* features = (const float*)d_in[1];
  const float* memory   = (const float*)d_in[2];
  const float* W_in     = (const float*)d_in[6];
  const float* b_in     = (const float*)d_in[7];
  const float* Wq       = (const float*)d_in[8];
  const float* bq       = (const float*)d_in[9];
  const float* Wk       = (const float*)d_in[10];
  const float* bk       = (const float*)d_in[11];
  const float* Wv       = (const float*)d_in[12];
  const float* bv       = (const float*)d_in[13];
  const float* ln_g     = (const float*)d_in[14];
  const float* ln_b     = (const float*)d_in[15];

  // workspace carve-up (~398 MB total)
  char* p = (char*)d_ws;
  float* emb_f = (float*)p; p += (size_t)NN * HD * 4;
  bf16*  emb_b = (bf16*)p;  p += (size_t)NN * HD * 2;
  bf16*  qb    = (bf16*)p;  p += (size_t)NN * HD * 2;
  bf16*  kb    = (bf16*)p;  p += (size_t)NN * HD * 2;
  bf16*  vt    = (bf16*)p;  p += (size_t)NN * HD * 2;   // V, transposed [HD][NN]
  float* att   = (float*)p; p += (size_t)NN * HD * 4;
  float* Mst   = (float*)p; p += (size_t)NN * 4;
  float* Linv  = (float*)p; p += (size_t)NN * 4;
  bf16*  wtin  = (bf16*)p;  p += (size_t)KIN * HD * 2;  // W_in^T bf16 [HD][KIN]
  bf16*  wtqkv[2][3];
  for (int l = 0; l < 2; l++)
    for (int m = 0; m < 3; m++) { wtqkv[l][m] = (bf16*)p; p += (size_t)HD * HD * 2; }
  float* S = (float*)p; p += (size_t)NN * NN * 4;

  // one-time weight transpose+convert to bf16 [n][k]
  k_wt<<<dim3(KIN / 32, HD / 32), 256, 0, stream>>>(W_in, wtin, KIN);
  for (int l = 0; l < 2; l++) {
    k_wt<<<dim3(HD / 32, HD / 32), 256, 0, stream>>>(Wq + (size_t)l * HD * HD,
                                                     wtqkv[l][0], HD);
    k_wt<<<dim3(HD / 32, HD / 32), 256, 0, stream>>>(Wk + (size_t)l * HD * HD,
                                                     wtqkv[l][1], HD);
    k_wt<<<dim3(HD / 32, HD / 32), 256, 0, stream>>>(Wv + (size_t)l * HD * HD,
                                                     wtqkv[l][2], HD);
  }

  dim3 gGemm(NN / 128, HD / 128);

  k_inproj<<<gGemm, 256, 0, stream>>>(features, memory, wtin, b_in, emb_f, emb_b);

  for (int l = 0; l < 2; l++) {
    k_proj<false><<<gGemm, 256, 0, stream>>>(emb_b, wtqkv[l][0],
                                             bq + (size_t)l * HD, qb);
    k_proj<false><<<gGemm, 256, 0, stream>>>(emb_b, wtqkv[l][1],
                                             bk + (size_t)l * HD, kb);
    k_proj<true><<<gGemm, 256, 0, stream>>>(emb_b, wtqkv[l][2],
                                            bv + (size_t)l * HD, vt);

    k_scores<<<NN / 64, 128, 0, stream>>>(qb, kb, S, Mst, Linv);
    k_pv<<<gGemm, 256, 0, stream>>>(S, Mst, Linv, vt, att);

    float* of = (l == 1) ? (float*)d_out : emb_f;
    k_ln<<<NN, 256, 0, stream>>>(emb_f, att, ln_g + (size_t)l * HD,
                                 ln_b + (size_t)l * HD, of, emb_b);
  }
}
